// MultiModelWithScalarHeads_49486613185071
// MI455X (gfx1250) — matrix operand order repaired and checked by reading, not compiled, not run on hardware
//
#include <hip/hip_runtime.h>
#include <hip/hip_bf16.h>
#include <stdint.h>

typedef __attribute__((ext_vector_type(16))) __bf16 v16bf;
typedef __attribute__((ext_vector_type(8)))  __bf16 v8bf;
typedef __attribute__((ext_vector_type(8)))  float  v8f;

// ---------------------------------------------------------------------------
// W [M,K,N] fp32  ->  Wt [M,N,K] bf16   (LDS tile transpose, 32x32 tiles)
// ---------------------------------------------------------------------------
__global__ __launch_bounds__(256)
void transpose_cvt_bf16(const float* __restrict__ W, __bf16* __restrict__ Wt,
                        int K, int N)
{
    __shared__ float t[32][33];
    const int nt = N >> 5, kt = K >> 5;
    const int id = blockIdx.x;
    const int n0 = (id % nt) << 5;
    const int k0 = ((id / nt) % kt) << 5;
    const int m  = id / (nt * kt);

    const float* Wb  = W  + (size_t)m * K * N;
    __bf16*      Wtb = Wt + (size_t)m * N * K;

    const int r = threadIdx.x >> 5;   // 0..7
    const int c = threadIdx.x & 31;   // 0..31
#pragma unroll
    for (int p = 0; p < 4; ++p)
        t[p * 8 + r][c] = Wb[(size_t)(k0 + p * 8 + r) * N + (n0 + c)];
    __syncthreads();
#pragma unroll
    for (int p = 0; p < 4; ++p)
        Wtb[(size_t)(n0 + p * 8 + r) * K + (k0 + c)] = (__bf16)t[c][p * 8 + r];
}

// ---------------------------------------------------------------------------
// GEMM with bias: Y[m,b,s,n] = sum_k A[m,b,s,k] * W[m,k,n] + bias[m,n]
// A fp32 (global), W pre-transposed bf16 Wt[m][n][k], Y fp32.
// Block tile 128(s) x 64(n), 8 waves, wave tile 16x64, K-step 64
// (2 WMMA K-substeps per staged tile -> 8 v_wmma per barrier interval).
// Double-buffered LDS staged with GLOBAL_LOAD_ASYNC_TO_LDS_B128 (ASYNCcnt).
// ---------------------------------------------------------------------------
#define A_ROWPAD 68                        // 64 fp32 + 4 pad (272 B rows, 16B mult)
#define B_ROWPAD 72                        // 64 bf16 + 8 pad (144 B rows, 16B mult)
#define A_BUFSTRIDE (128 * A_ROWPAD * 4)   // 34816 B
#define B_BUFSTRIDE (64 * B_ROWPAD * 2)    // 9216 B

__global__ __launch_bounds__(256)
void gemm_bias_wmma(const float* __restrict__ A, const __bf16* __restrict__ Wt,
                    const float* __restrict__ bias, float* __restrict__ Y,
                    int S, int K, int N, int Bb, int ntiles, int stiles)
{
    __shared__ float  Afp[2][128][A_ROWPAD];  // fp32 A tile 128 x 64
    __shared__ __bf16 Bbf[2][64][B_ROWPAD];   // bf16 B tile [n][k] 64 x 64

    const int id = blockIdx.x;
    const int ft = id % ntiles;
    const int st = (id / ntiles) % stiles;
    const int b  = (id / (ntiles * stiles)) % Bb;
    const int m  =  id / (ntiles * stiles * Bb);

    const int s0 = st * 128;
    const int f0 = ft * 64;

    const float*  Ab  = A  + ((size_t)(m * Bb + b) * S + s0) * K;
    const __bf16* Wtb = Wt + ((size_t)m * N + f0) * K;
    float*        Yb  = Y  + (size_t)(m * Bb + b) * S * N;

    const int tid  = threadIdx.x;
    const int lane = tid & 31;
    const int wv   = tid >> 5;

    // staging assignment (per thread)
    const int ar = tid >> 4;          // A row within 16-row pass (0..15)
    const int ac = (tid & 15) << 2;   // A col (floats): 0..60
    // B: 64 rows x 64 k bf16 = 512 x 16B; issue i = p*256 + tid
    //    n = i >> 3, ko = (i & 7) * 8

    // buffer-0 LDS byte addresses (low 32 bits of flat ptr == LDS offset);
    // buffer 1 = base + stride (pure arithmetic, avoids v_movrels indexing)
    uint32_t a0[8], b0[2];
#pragma unroll
    for (int p = 0; p < 8; ++p)
        a0[p] = (uint32_t)(uintptr_t)&Afp[0][p * 16 + ar][ac];
#pragma unroll
    for (int p = 0; p < 2; ++p) {
        const int i = p * 256 + tid;
        b0[p] = (uint32_t)(uintptr_t)&Bbf[0][i >> 3][(i & 7) << 3];
    }

    auto issue_tile = [&](int bf, int k0) {
        const uint32_t aoff = bf ? (uint32_t)A_BUFSTRIDE : 0u;
        const uint32_t boff = bf ? (uint32_t)B_BUFSTRIDE : 0u;
#pragma unroll
        for (int p = 0; p < 8; ++p) {
            const float* g = Ab + (size_t)(p * 16 + ar) * K + (k0 + ac);
            asm volatile("global_load_async_to_lds_b128 %0, %1, off"
                         :: "v"(a0[p] + aoff), "v"((uint64_t)(uintptr_t)g)
                         : "memory");
        }
#pragma unroll
        for (int p = 0; p < 2; ++p) {
            const int i = p * 256 + tid;
            const __bf16* g = Wtb + (size_t)(i >> 3) * K + (k0 + ((i & 7) << 3));
            asm volatile("global_load_async_to_lds_b128 %0, %1, off"
                         :: "v"(b0[p] + boff), "v"((uint64_t)(uintptr_t)g)
                         : "memory");
        }
    };

    issue_tile(0, 0);

    v8f acc[4];
#pragma unroll
    for (int c = 0; c < 4; ++c)
#pragma unroll
        for (int j = 0; j < 8; ++j) acc[c][j] = 0.0f;

    const int rowA = wv * 16 + (lane & 15);
    const int kbA  = (lane >> 4) << 3;  // 0 or 8  (A frag: K 0..7/16..23 vs 8..15/24..31)
    const int kbB  = (lane >> 4) << 3;  // 0 or 8  (B frag: K 0..7/16..23 vs 8..15/24..31)
    const int nrow = lane & 15;

    int buf = 0;
    for (int k0 = 0; k0 < K; k0 += 64, buf ^= 1) {
        asm volatile("s_wait_asynccnt 0x0" ::: "memory");
        __syncthreads();
        if (k0 + 64 < K) issue_tile(buf ^ 1, k0 + 64);

        const float* ap = &Afp[buf][rowA][0];

#pragma unroll
        for (int kk = 0; kk < 64; kk += 32) {
            // ---- A fragment: fp32 LDS -> bf16 regs (ISA 16-bit A layout) ----
            const float4 fa0 = *(const float4*)(ap + kk + kbA);
            const float4 fa1 = *(const float4*)(ap + kk + kbA + 4);
            const float4 fa2 = *(const float4*)(ap + kk + kbA + 16);
            const float4 fa3 = *(const float4*)(ap + kk + kbA + 20);
            v16bf a;
            a[0]  = (__bf16)fa0.x; a[1]  = (__bf16)fa0.y; a[2]  = (__bf16)fa0.z; a[3]  = (__bf16)fa0.w;
            a[4]  = (__bf16)fa1.x; a[5]  = (__bf16)fa1.y; a[6]  = (__bf16)fa1.z; a[7]  = (__bf16)fa1.w;
            a[8]  = (__bf16)fa2.x; a[9]  = (__bf16)fa2.y; a[10] = (__bf16)fa2.z; a[11] = (__bf16)fa2.w;
            a[12] = (__bf16)fa3.x; a[13] = (__bf16)fa3.y; a[14] = (__bf16)fa3.z; a[15] = (__bf16)fa3.w;

            // ---- load all 4 B fragments first (batch the DS traffic) ----
            v16bf bbv[4];
#pragma unroll
            for (int c = 0; c < 4; ++c) {
                const __bf16* bp = &Bbf[buf][c * 16 + nrow][kk + kbB];
                const v8bf blo = *(const v8bf*)bp;
                const v8bf bhi = *(const v8bf*)(bp + 16);
                bbv[c] = __builtin_shufflevector(blo, bhi,
                    0, 1, 2, 3, 4, 5, 6, 7, 8, 9, 10, 11, 12, 13, 14, 15);
            }
            // ---- then the 4 WMMAs ----
#pragma unroll
            for (int c = 0; c < 4; ++c) {
                acc[c] = __builtin_amdgcn_wmma_f32_16x16x32_bf16(
                    false, a, false, bbv[c], (short)0, acc[c], false, false);
            }
        }
        // single barrier per iteration (buffer reuse protected by next barrier)
    }

    // ---- epilogue: C/D layout -> global, add bias ----
    const int rr = wv * 16 + ((lane >> 4) << 3);
#pragma unroll
    for (int c = 0; c < 4; ++c) {
        const int n = f0 + c * 16 + nrow;
        const float bv = bias[(size_t)m * N + n];
#pragma unroll
        for (int j = 0; j < 8; ++j) {
            Yb[(size_t)(s0 + rr + j) * N + n] = acc[c][j] + bv;
        }
    }
}

// ---------------------------------------------------------------------------
// BatchNorm over the batch axis (B=4) per (m, s, f), + ReLU, in place.
// ---------------------------------------------------------------------------
__global__ __launch_bounds__(256)
void bn_relu_inplace(float* __restrict__ Y, const float* __restrict__ g,
                     const float* __restrict__ be, int SF, int F)
{
    const long i = (long)blockIdx.x * 256 + threadIdx.x;  // over M*S*F
    const int m  = (int)(i / SF);
    const int sf = (int)(i % SF);
    const int f  = sf % F;

    float* p = Y + (size_t)m * 4 * SF + sf;
    const float y0 = p[0];
    const float y1 = p[(size_t)SF];
    const float y2 = p[(size_t)2 * SF];
    const float y3 = p[(size_t)3 * SF];

    const float mean = (y0 + y1 + y2 + y3) * 0.25f;
    const float d0 = y0 - mean, d1 = y1 - mean, d2 = y2 - mean, d3 = y3 - mean;
    const float var = (d0 * d0 + d1 * d1 + d2 * d2 + d3 * d3) * 0.25f;
    const float inv = rsqrtf(var + 1e-5f);

    const float gg = g[(size_t)m * F + f];
    const float bb = be[(size_t)m * F + f];

    p[0]              = fmaxf(0.0f, gg * d0 * inv + bb);
    p[(size_t)SF]     = fmaxf(0.0f, gg * d1 * inv + bb);
    p[(size_t)2 * SF] = fmaxf(0.0f, gg * d2 * inv + bb);
    p[(size_t)3 * SF] = fmaxf(0.0f, gg * d3 * inv + bb);
}

// ---------------------------------------------------------------------------
// Fused BN2 + ReLU + W3 dot + b3 -> out[b, s, m].  One wave per (m, s).
// ---------------------------------------------------------------------------
__global__ __launch_bounds__(256)
void bn_head(const float* __restrict__ Y2, const float* __restrict__ g2,
             const float* __restrict__ be2, const float* __restrict__ W3,
             const float* __restrict__ b3, float* __restrict__ out,
             int S, int G, int M)
{
    const int lane = threadIdx.x & 31;
    const int wv   = threadIdx.x >> 5;
    const long ms  = (long)blockIdx.x * 8 + wv;   // over M*S
    if (ms >= (long)M * S) return;
    const int m = (int)(ms / S);
    const int s = (int)(ms % S);

    const float* base = Y2 + ((size_t)m * 4 * S + s) * G;

    float acc[4] = {0.0f, 0.0f, 0.0f, 0.0f};
#pragma unroll
    for (int j = 0; j < 4; ++j) {
        const int gidx = lane + 32 * j;
        const float y0 = base[gidx];
        const float y1 = base[(size_t)1 * S * G + gidx];
        const float y2 = base[(size_t)2 * S * G + gidx];
        const float y3 = base[(size_t)3 * S * G + gidx];

        const float mean = (y0 + y1 + y2 + y3) * 0.25f;
        const float d0 = y0 - mean, d1 = y1 - mean, d2 = y2 - mean, d3 = y3 - mean;
        const float var = (d0 * d0 + d1 * d1 + d2 * d2 + d3 * d3) * 0.25f;
        const float inv = rsqrtf(var + 1e-5f);

        const float gg = g2[(size_t)m * G + gidx];
        const float bb = be2[(size_t)m * G + gidx];
        const float w  = W3[(size_t)m * G + gidx];

        acc[0] += fmaxf(0.0f, gg * d0 * inv + bb) * w;
        acc[1] += fmaxf(0.0f, gg * d1 * inv + bb) * w;
        acc[2] += fmaxf(0.0f, gg * d2 * inv + bb) * w;
        acc[3] += fmaxf(0.0f, gg * d3 * inv + bb) * w;
    }
#pragma unroll
    for (int off = 16; off > 0; off >>= 1) {
#pragma unroll
        for (int b = 0; b < 4; ++b) acc[b] += __shfl_xor(acc[b], off, 32);
    }
    if (lane < 4) {
        out[((size_t)lane * S + s) * M + m] = acc[lane] + b3[m];  // out [B,S,M]
    }
}

extern "C" void kernel_launch(void* const* d_in, const int* in_sizes, int n_in,
                              void* d_out, int out_size, void* d_ws, size_t ws_size,
                              hipStream_t stream) {
    (void)in_sizes; (void)n_in; (void)out_size; (void)ws_size;

    const int M = 2, B = 4, S = 4096, H = 4096, F = 256, G = 128;

    const float* hs  = (const float*)d_in[0];
    const float* W1  = (const float*)d_in[1];
    const float* b1  = (const float*)d_in[2];
    const float* g1  = (const float*)d_in[3];
    const float* be1 = (const float*)d_in[4];
    const float* W2  = (const float*)d_in[5];
    const float* b2  = (const float*)d_in[6];
    const float* g2  = (const float*)d_in[7];
    const float* be2 = (const float*)d_in[8];
    const float* W3  = (const float*)d_in[9];
    const float* b3  = (const float*)d_in[10];
    float* out = (float*)d_out;

    // workspace layout
    float*  Y1  = (float*)d_ws;                          // [M,B,S,F] 32 MiB
    float*  Y2  = Y1 + (size_t)M * B * S * F;            // [M,B,S,G] 16 MiB
    __bf16* Wt1 = (__bf16*)(Y2 + (size_t)M * B * S * G); // [M,F,H] bf16 4 MiB
    __bf16* Wt2 = Wt1 + (size_t)M * F * H;               // [M,G,F] bf16 128 KiB

    // W1 [M,H,F] -> Wt1 [M,F,H] bf16 ; W2 [M,F,G] -> Wt2 [M,G,F] bf16
    transpose_cvt_bf16<<<dim3(M * (H / 32) * (F / 32)), 256, 0, stream>>>(W1, Wt1, H, F);
    transpose_cvt_bf16<<<dim3(M * (F / 32) * (G / 32)), 256, 0, stream>>>(W2, Wt2, F, G);

    // GEMM1: hs x W1 + b1 -> Y1
    {
        const int stiles = S / 128, ntiles = F / 64;
        gemm_bias_wmma<<<dim3(M * B * stiles * ntiles), 256, 0, stream>>>(
            hs, Wt1, b1, Y1, S, H, F, B, ntiles, stiles);
    }
    // BN1 + ReLU in place
    {
        const long total = (long)M * S * F;
        bn_relu_inplace<<<dim3((unsigned)(total / 256)), 256, 0, stream>>>(Y1, g1, be1, S * F, F);
    }
    // GEMM2: Y1 x W2 + b2 -> Y2
    {
        const int stiles = S / 128, ntiles = G / 64;
        gemm_bias_wmma<<<dim3(M * B * stiles * ntiles), 256, 0, stream>>>(
            Y1, Wt2, b2, Y2, S, F, G, B, ntiles, stiles);
    }
    // BN2 + ReLU + W3 + b3 -> out
    {
        bn_head<<<dim3((unsigned)((long)M * S / 8)), 256, 0, stream>>>(
            Y2, g2, be2, W3, b3, out, S, G, M);
    }
}
